// DiffusionGATNetwork_60069412602240
// MI455X (gfx1250) — matrix operand order repaired and checked by reading, not compiled, not run on hardware
//
#include <hip/hip_runtime.h>
#include <hip/hip_bf16.h>
#include <float.h>

// ---------------- problem constants ----------------
#define BB     16
#define LL     200
#define LP1    201
#define UU     30000
#define NINP   120
#define NT     8
#define DD     128
#define NH     8
#define DK     16
#define NROWS  (BB*LL)            // 3200
#define NEGF   (-4294967295.0f)   // -2^32 + 1

typedef __attribute__((ext_vector_type(16))) _Float16 v16h;
typedef __attribute__((ext_vector_type(8)))  _Float16 v8h;
typedef __attribute__((ext_vector_type(8)))  float    v8f;

// ---------------- workspace layout (bytes) ----------------
#define OFF_IDX       ((size_t)0)                 // int[256]
#define OFF_LIVE      ((size_t)1024)              // float[256]
#define OFF_DYTS      ((size_t)2048)              // int[256]
#define OFF_INTERVAL  ((size_t)4096)              // 200*128 f32   = 102,400
#define OFF_CASEMB    ((size_t)131072)            // 3200*128 f32  = 1,638,400
#define OFF_ALPHA     ((size_t)1835008)           // 16*200*200 f32= 2,560,000
#define OFF_SEQ       ((size_t)4456448)           // 3200*128 f32
#define OFF_Q         ((size_t)6291456)
#define OFF_K         ((size_t)8388608)
#define OFF_V         ((size_t)10485760)
#define OFF_CTX       ((size_t)12582912)
#define OFF_X         ((size_t)14680064)
#define OFF_X2H       ((size_t)16777216)          // 3200*128 f16  = 819,200
#define OFF_WH        ((size_t)18874368)          // 30000*128 f16 = 7,680,000

// =====================================================================
// K0: ts_max, stopped, searchsorted idx, live, dy_ts, interval_embs
// =====================================================================
__global__ void prep_kernel(const float* __restrict__ cas_tss,
                            const float* __restrict__ dyn_times,
                            const float* __restrict__ time_emb_w,
                            int* __restrict__ idx_l, float* __restrict__ livef,
                            int* __restrict__ dyts, float* __restrict__ interval) {
  __shared__ float tsmax[LL];
  __shared__ int   stop[LL];
  __shared__ int   dy[LL];
  const int t = threadIdx.x;
  if (t < LL) {
    float m = -FLT_MAX;
    for (int b = 0; b < BB; ++b) m = fmaxf(m, cas_tss[b * LP1 + t]);
    tsmax[t] = m;
  }
  __syncthreads();
  if (t == 0) {
    int s = 0;
    for (int l = 0; l < LL; ++l) { if (tsmax[l] < 1.0f) s = 1; stop[l] = s; }
  }
  __syncthreads();
  if (t < LL) {
    float v = tsmax[t];
    int idx = -1;
    for (int i = 0; i < NT; ++i) if (dyn_times[i] <= v) idx = i;   // searchsorted right - 1
    if (idx < 0) idx = NT - 1;
    const int st = stop[t];
    idx_l[t] = idx;
    livef[t] = st ? 0.0f : 1.0f;
    const int d = st ? 0 : idx;
    dyts[t] = d;
    dy[t] = d;
  }
  __syncthreads();
  for (int i = t; i < LL * DD; i += blockDim.x) {
    const int l = i / DD, d = i % DD;
    interval[i] = time_emb_w[dy[l] * DD + d];
  }
}

// =====================================================================
// K1: cas_embs = concat(dyn_embs[idx,uid]*live, pos_emb)
// =====================================================================
__global__ void embed_kernel(const int* __restrict__ cas_uids,
                             const float* __restrict__ dyn_embs,
                             const float* __restrict__ pos_emb_w,
                             const int* __restrict__ idx_l,
                             const float* __restrict__ livef,
                             float* __restrict__ casemb) {
  const int row = blockIdx.x;            // b*L + l
  const int b = row / LL, l = row % LL;
  const int d = threadIdx.x;             // 0..127
  float v;
  if (d < NINP) {
    const int uid = cas_uids[b * LP1 + l];
    v = dyn_embs[((long)idx_l[l] * UU + uid) * NINP + d] * livef[l];
  } else {
    v = pos_emb_w[l * 8 + (d - NINP)];
  }
  casemb[(long)row * DD + d] = v;
}

// =====================================================================
// K2: affine + mask + softmax over q (axis=1)  -> alpha[b,q,k]
// one block per (b,k) column
// =====================================================================
__global__ void col_softmax_kernel(const int* __restrict__ cas_uids,
                                   const float* __restrict__ casemb,
                                   const float* __restrict__ interval,
                                   float* __restrict__ alpha) {
  const int b = blockIdx.x, k = blockIdx.y;
  __shared__ float tk[DD];
  __shared__ float red[256];
  __shared__ float sc[256];
  const int t = threadIdx.x;
  for (int i = t; i < DD; i += 256) tk[i] = interval[k * DD + i];
  __syncthreads();
  float s = -FLT_MAX;
  if (t < LL) {
    const bool mask = (k > t) || (cas_uids[b * LP1 + t] == 0);
    if (mask) {
      s = NEGF;
    } else {
      float acc = 0.f;
      const float* cr = casemb + ((long)(b * LL + t)) * DD;
      for (int j = 0; j < DD; ++j) acc += cr[j] * tk[j];
      s = acc / (sqrtf(128.0f) + 1e-6f);
    }
    sc[t] = s;
  }
  red[t] = (t < LL) ? s : -FLT_MAX;
  __syncthreads();
  for (int off = 128; off > 0; off >>= 1) {
    if (t < off) red[t] = fmaxf(red[t], red[t + off]);
    __syncthreads();
  }
  const float mx = red[0];
  __syncthreads();
  float e = 0.f;
  if (t < LL) { e = expf(sc[t] - mx); sc[t] = e; }
  red[t] = (t < LL) ? e : 0.f;
  __syncthreads();
  for (int off = 128; off > 0; off >>= 1) {
    if (t < off) red[t] += red[t + off];
    __syncthreads();
  }
  const float inv = 1.0f / red[0];
  if (t < LL) alpha[((long)(b * LL + t)) * LL + k] = sc[t] * inv;
}

// =====================================================================
// K3: seq[b,q,:] = sum_k alpha[b,q,k] * casemb[b,k,:]
// =====================================================================
__global__ void seq_kernel(const float* __restrict__ alpha,
                           const float* __restrict__ casemb,
                           float* __restrict__ seq) {
  const int row = blockIdx.x;            // b*L + q
  const int b = row / LL;
  __shared__ float al[LL];
  const int t = threadIdx.x;             // 128
  for (int i = t; i < LL; i += 128) al[i] = alpha[(long)row * LL + i];
  __syncthreads();
  float acc = 0.f;
  for (int k = 0; k < LL; ++k) acc += al[k] * casemb[((long)(b * LL + k)) * DD + t];
  seq[(long)row * DD + t] = acc;
}

// =====================================================================
// K4: row-wise 128x128 projection (used for Wq, Wk, Wv)
// =====================================================================
__global__ void proj_kernel(const float* __restrict__ in,
                            const float* __restrict__ W,
                            float* __restrict__ out) {
  const int row = blockIdx.x;
  __shared__ float r[DD];
  const int t = threadIdx.x;
  r[t] = in[(long)row * DD + t];
  __syncthreads();
  float acc = 0.f;
  for (int j = 0; j < DD; ++j) acc += r[j] * W[j * DD + t];
  out[(long)row * DD + t] = acc;
}

// =====================================================================
// K5: per-head causal attention; one block per (q, h, b)
// =====================================================================
__global__ void attn_kernel(const int* __restrict__ cas_uids,
                            const float* __restrict__ q,
                            const float* __restrict__ k,
                            const float* __restrict__ v,
                            float* __restrict__ ctx) {
  const int qq = blockIdx.x, h = blockIdx.y, b = blockIdx.z;
  __shared__ float al[256];
  __shared__ float red[256];
  __shared__ float qv[DK];
  const int t = threadIdx.x;
  if (t < DK) qv[t] = q[((long)(b * LL + qq)) * DD + h * DK + t];
  __syncthreads();
  float s = -FLT_MAX;
  if (t < LL) {
    const bool mask = (t > qq) || (cas_uids[b * LP1 + t] == 0);
    if (mask) {
      s = NEGF;
    } else {
      float acc = 0.f;
      const float* kr = k + ((long)(b * LL + t)) * DD + h * DK;
      for (int d = 0; d < DK; ++d) acc += qv[d] * kr[d];
      s = acc * 0.25f;                  // 1/sqrt(dk), dk=16
    }
    al[t] = s;
  }
  red[t] = (t < LL) ? s : -FLT_MAX;
  __syncthreads();
  for (int off = 128; off > 0; off >>= 1) {
    if (t < off) red[t] = fmaxf(red[t], red[t + off]);
    __syncthreads();
  }
  const float mx = red[0];
  __syncthreads();
  float e = 0.f;
  if (t < LL) { e = expf(al[t] - mx); al[t] = e; }
  red[t] = (t < LL) ? e : 0.f;
  __syncthreads();
  for (int off = 128; off > 0; off >>= 1) {
    if (t < off) red[t] += red[t + off];
    __syncthreads();
  }
  const float inv = 1.0f / red[0];
  __syncthreads();
  if (t < LL) al[t] *= inv;
  __syncthreads();
  if (t < DK) {
    float acc = 0.f;
    for (int kk = 0; kk < LL; ++kk)
      acc += al[kk] * v[((long)(b * LL + kk)) * DD + h * DK + t];
    ctx[((long)(b * LL + qq)) * DD + h * DK + t] = acc;
  }
}

// =====================================================================
// K6: x = LN(seq + ctx @ Wo)
// =====================================================================
__global__ void oproj_ln_kernel(const float* __restrict__ seq,
                                const float* __restrict__ ctx,
                                const float* __restrict__ Wo,
                                const float* __restrict__ g,
                                const float* __restrict__ bta,
                                float* __restrict__ x) {
  const int row = blockIdx.x;
  __shared__ float cr[DD];
  __shared__ float red[DD];
  const int t = threadIdx.x;
  cr[t] = ctx[(long)row * DD + t];
  __syncthreads();
  float acc = 0.f;
  for (int j = 0; j < DD; ++j) acc += cr[j] * Wo[j * DD + t];
  const float val = seq[(long)row * DD + t] + acc;
  red[t] = val;
  __syncthreads();
  for (int off = 64; off > 0; off >>= 1) {
    if (t < off) red[t] += red[t + off];
    __syncthreads();
  }
  const float mean = red[0] * (1.0f / DD);
  __syncthreads();
  const float dv = val - mean;
  red[t] = dv * dv;
  __syncthreads();
  for (int off = 64; off > 0; off >>= 1) {
    if (t < off) red[t] += red[t + off];
    __syncthreads();
  }
  const float var = red[0] * (1.0f / DD);
  x[(long)row * DD + t] = dv * rsqrtf(var + 1e-5f) * g[t] + bta[t];
}

// =====================================================================
// K7: x2 = LN(x + relu(x@W1+b1)@W2+b2)   -> f16 output for WMMA GEMM
// =====================================================================
__global__ void ffn_ln_kernel(const float* __restrict__ x,
                              const float* __restrict__ W1,
                              const float* __restrict__ b1,
                              const float* __restrict__ W2,
                              const float* __restrict__ b2,
                              const float* __restrict__ g,
                              const float* __restrict__ bta,
                              _Float16* __restrict__ x2h) {
  const int row = blockIdx.x;
  __shared__ float xr[DD];
  __shared__ float hr[DD];
  __shared__ float red[DD];
  const int t = threadIdx.x;
  const float xv = x[(long)row * DD + t];
  xr[t] = xv;
  __syncthreads();
  float acc = 0.f;
  for (int j = 0; j < DD; ++j) acc += xr[j] * W1[j * DD + t];
  hr[t] = fmaxf(acc + b1[t], 0.0f);
  __syncthreads();
  acc = 0.f;
  for (int j = 0; j < DD; ++j) acc += hr[j] * W2[j * DD + t];
  const float val = xv + acc + b2[t];
  red[t] = val;
  __syncthreads();
  for (int off = 64; off > 0; off >>= 1) {
    if (t < off) red[t] += red[t + off];
    __syncthreads();
  }
  const float mean = red[0] * (1.0f / DD);
  __syncthreads();
  const float dv = val - mean;
  red[t] = dv * dv;
  __syncthreads();
  for (int off = 64; off > 0; off >>= 1) {
    if (t < off) red[t] += red[t + off];
    __syncthreads();
  }
  const float var = red[0] * (1.0f / DD);
  const float y = dv * rsqrtf(var + 1e-5f) * g[t] + bta[t];
  x2h[(long)row * DD + t] = (_Float16)y;
}

// =====================================================================
// K8: f32 -> f16 conversion (fc_w)
// =====================================================================
__global__ void h_convert_kernel(const float* __restrict__ w,
                                 _Float16* __restrict__ wh, long n) {
  const long i = (long)blockIdx.x * blockDim.x + threadIdx.x;
  if (i < n) wh[i] = (_Float16)w[i];
}

// =====================================================================
// K9: final FC via WMMA:  out[3200,30000] = x2h[3200,128] * fc_w^T + fc_b
//
// Block tile 128(M) x 240(N): 8 waves, each owns 16 rows and all 15 N
// sub-tiles (60 v_wmma_f32_16x16x32_f16 per wave). The 240x128 f16 B tile
// (fc_w rows n0..n0+239) is staged once per block into LDS with the CDNA5
// async copy path (global_load_async_to_lds_b128, drained by
// s_wait_asynccnt), cutting L2 read traffic ~6x vs per-wave B loads.
// LDS rows padded to 136 halfs so B-fragment ds_load_b128 spreads over all
// 64 banks.
// =====================================================================
#define BN     240
#define BROWH  136   // padded halfs per LDS B row (272 B)

__global__ void __launch_bounds__(256)
fc_wmma_kernel(const _Float16* __restrict__ xh,
               const _Float16* __restrict__ wh,
               const float* __restrict__ fc_b,
               float* __restrict__ out) {
  __shared__ _Float16 ldsB[BN * BROWH];   // 65,280 bytes
  const int m0 = blockIdx.x * 128;
  const int n0 = blockIdx.y * BN;
  const int tid = threadIdx.x;

  // ---- async copy B tile (240 rows x 128 halfs) global -> LDS ----
  for (int c = tid; c < BN * 16; c += 256) {
    const int row = c >> 4;               // 0..239
    const int c16 = c & 15;               // 16-byte chunk within row
    const unsigned ldsaddr =
        (unsigned)(uintptr_t)(&ldsB[row * BROWH + c16 * 8]);
    const _Float16* gp = wh + (size_t)(n0 + row) * DD + c16 * 8;
    asm volatile("global_load_async_to_lds_b128 %0, %1, off"
                 :: "v"(ldsaddr), "v"(gp)
                 : "memory");
  }
  asm volatile("s_wait_asynccnt 0x0" ::: "memory");
  __syncthreads();

  const int wave = tid >> 5;
  const int lane = tid & 31;
  const int hi = lane >> 4;
  const int lm = lane & 15;

  v8f acc[15];
#pragma unroll
  for (int t = 0; t < 15; ++t) acc[t] = (v8f){};

  const int arow = m0 + wave * 16 + lm;   // A: M = lane%16
  const int aklo = hi * 8;                // A 16x32 f16 per-lane K base

  for (int kk = 0; kk < DD; kk += 32) {
    // A fragment: halves 0..7 -> K=aklo.., halves 8..15 -> K=aklo+16..
    const _Float16* ap = xh + (size_t)arow * DD + kk + aklo;
    const v8h a0 = *(const v8h*)(ap);
    const v8h a1 = *(const v8h*)(ap + 16);
    v16h A;
#pragma unroll
    for (int i = 0; i < 8; ++i) { A[i] = a0[i]; A[i + 8] = a1[i]; }

    const int bk = kk + hi * 8;           // B: halves 0..7 -> K=bk.., halves 8..15 -> K=bk+16..
#pragma unroll
    for (int t = 0; t < 15; ++t) {
      const _Float16* bp = &ldsB[(t * 16 + lm) * BROWH + bk];
      const v8h b0 = *(const v8h*)(bp);
      const v8h b1 = *(const v8h*)(bp + 16);
      v16h Bf;
#pragma unroll
      for (int i = 0; i < 8; ++i) { Bf[i] = b0[i]; Bf[i + 8] = b1[i]; }
      acc[t] = __builtin_amdgcn_wmma_f32_16x16x32_f16(
          false, A, false, Bf, (short)0, acc[t], false, false);
    }
  }

  // ---- store C tile + bias (VGPR j -> M = j + 8*(lane/16), N = lane%16)
#pragma unroll
  for (int t = 0; t < 15; ++t) {
    const int n = n0 + t * 16 + lm;
    const float bias = fc_b[n];
#pragma unroll
    for (int j = 0; j < 8; ++j) {
      const int m = m0 + wave * 16 + j + hi * 8;
      out[(long)m * UU + n] = acc[t][j] + bias;
    }
  }
}

// =====================================================================
// K10: scatter -inf mask: out[b,l,0] and out[b,l,uids[b,k]] for k<=l
// =====================================================================
__global__ void neg_mask_kernel(const int* __restrict__ cas_uids,
                                float* __restrict__ out) {
  const int l = blockIdx.x, b = blockIdx.y;
  const long row = (long)(b * LL + l);
  const int t = threadIdx.x;
  const float ninf = -__builtin_inff();
  if (t == 0) out[row * UU] = ninf;
  for (int k = t; k <= l; k += blockDim.x) {
    const int u = cas_uids[b * LP1 + k];
    out[row * UU + u] = ninf;
  }
}

// =====================================================================
extern "C" void kernel_launch(void* const* d_in, const int* in_sizes, int n_in,
                              void* d_out, int out_size, void* d_ws, size_t ws_size,
                              hipStream_t stream) {
  const int*   cas_uids   = (const int*)  d_in[0];
  const float* cas_tss    = (const float*)d_in[1];
  const float* dyn_times  = (const float*)d_in[2];
  const float* dyn_embs   = (const float*)d_in[3];
  const float* time_emb_w = (const float*)d_in[4];
  const float* pos_emb_w  = (const float*)d_in[5];
  const float* Wq         = (const float*)d_in[6];
  const float* Wk         = (const float*)d_in[7];
  const float* Wv         = (const float*)d_in[8];
  const float* Wo         = (const float*)d_in[9];
  const float* ln1_g      = (const float*)d_in[10];
  const float* ln1_b      = (const float*)d_in[11];
  const float* ffn_w1     = (const float*)d_in[12];
  const float* ffn_b1     = (const float*)d_in[13];
  const float* ffn_w2     = (const float*)d_in[14];
  const float* ffn_b2     = (const float*)d_in[15];
  const float* ln2_g      = (const float*)d_in[16];
  const float* ln2_b      = (const float*)d_in[17];
  const float* fc_w       = (const float*)d_in[18];
  const float* fc_b       = (const float*)d_in[19];
  float* out = (float*)d_out;

  char* ws = (char*)d_ws;
  int*      ws_idx      = (int*)   (ws + OFF_IDX);
  float*    ws_live     = (float*) (ws + OFF_LIVE);
  int*      ws_dyts     = (int*)   (ws + OFF_DYTS);
  float*    ws_interval = (float*) (ws + OFF_INTERVAL);
  float*    ws_casemb   = (float*) (ws + OFF_CASEMB);
  float*    ws_alpha    = (float*) (ws + OFF_ALPHA);
  float*    ws_seq      = (float*) (ws + OFF_SEQ);
  float*    ws_q        = (float*) (ws + OFF_Q);
  float*    ws_k        = (float*) (ws + OFF_K);
  float*    ws_v        = (float*) (ws + OFF_V);
  float*    ws_ctx      = (float*) (ws + OFF_CTX);
  float*    ws_x        = (float*) (ws + OFF_X);
  _Float16* ws_x2h      = (_Float16*)(ws + OFF_X2H);
  _Float16* ws_wh       = (_Float16*)(ws + OFF_WH);

  prep_kernel<<<1, 256, 0, stream>>>(cas_tss, dyn_times, time_emb_w,
                                     ws_idx, ws_live, ws_dyts, ws_interval);
  embed_kernel<<<NROWS, 128, 0, stream>>>(cas_uids, dyn_embs, pos_emb_w,
                                          ws_idx, ws_live, ws_casemb);
  col_softmax_kernel<<<dim3(BB, LL), 256, 0, stream>>>(cas_uids, ws_casemb,
                                                       ws_interval, ws_alpha);
  seq_kernel<<<NROWS, 128, 0, stream>>>(ws_alpha, ws_casemb, ws_seq);
  proj_kernel<<<NROWS, 128, 0, stream>>>(ws_seq, Wq, ws_q);
  proj_kernel<<<NROWS, 128, 0, stream>>>(ws_seq, Wk, ws_k);
  proj_kernel<<<NROWS, 128, 0, stream>>>(ws_seq, Wv, ws_v);
  attn_kernel<<<dim3(LL, NH, BB), 256, 0, stream>>>(cas_uids, ws_q, ws_k, ws_v, ws_ctx);
  oproj_ln_kernel<<<NROWS, 128, 0, stream>>>(ws_seq, ws_ctx, Wo, ln1_g, ln1_b, ws_x);
  ffn_ln_kernel<<<NROWS, 128, 0, stream>>>(ws_x, ffn_w1, ffn_b1, ffn_w2, ffn_b2,
                                           ln2_g, ln2_b, ws_x2h);
  const long wn = (long)UU * DD;
  h_convert_kernel<<<(int)((wn + 255) / 256), 256, 0, stream>>>(fc_w, ws_wh, wn);
  fc_wmma_kernel<<<dim3(NROWS / 128, UU / BN), 256, 0, stream>>>(ws_x2h, ws_wh, fc_b, out);
  neg_mask_kernel<<<dim3(LL, BB), 256, 0, stream>>>(cas_uids, out);
}
